// DCM_SSMEncoder_49598282334580
// MI455X (gfx1250) — matrix operand order repaired and checked by reading, not compiled, not run on hardware
//
#include <hip/hip_runtime.h>
#include <math.h>

// ---------------- problem constants ----------------
constexpr int BB = 2;
constexpr int LL = 256;
constexpr int DD = 3584;
constexpr int NN = 64;
constexpr int MM = 64;
constexpr int CC = 256;

// ---------------- WMMA types ----------------
typedef __attribute__((ext_vector_type(16))) __bf16 v16bf;
typedef __attribute__((ext_vector_type(8)))  __bf16 v8bf;
typedef __attribute__((ext_vector_type(8)))  float  v8f;

// hardware f32 -> bf16 (clang emits v_cvt_pk_bf16_f32 for pairs)
static __device__ __forceinline__ __bf16 f2bf(float f) {
  return static_cast<__bf16>(f);
}

// =====================================================================
// streaming f32 -> bf16 convert (8 elems/thread, fully coalesced)
// n must be a multiple of 2048 (all our sizes are)
// =====================================================================
__global__ __launch_bounds__(256) void cvt_f32_bf16_kernel(
    const float* __restrict__ src, __bf16* __restrict__ dst)
{
  const size_t i = ((size_t)blockIdx.x * 256 + threadIdx.x) * 8;
  float4 a = *reinterpret_cast<const float4*>(src + i);
  float4 b = *reinterpret_cast<const float4*>(src + i + 4);
  v8bf o;
  o[0]=f2bf(a.x); o[1]=f2bf(a.y); o[2]=f2bf(a.z); o[3]=f2bf(a.w);
  o[4]=f2bf(b.x); o[5]=f2bf(b.y); o[6]=f2bf(b.z); o[7]=f2bf(b.w);
  *reinterpret_cast<v8bf*>(dst + i) = o;
}

// A fragment (16x32 bf16, wave32): lane holds row = lane&15,
// K = kc + (lane>=16 ? 8 : 0) + {0..7} and the same +16.
static __device__ __forceinline__ v16bf load_frag_a(const __bf16* __restrict__ p,
                                                    int ka) {
  v8bf lo = *reinterpret_cast<const v8bf*>(p + ka);
  v8bf hi = *reinterpret_cast<const v8bf*>(p + ka + 16);
  return __builtin_shufflevector(lo, hi, 0,1,2,3,4,5,6,7,8,9,10,11,12,13,14,15);
}

// B fragment (32x16 bf16): lane holds col = lane&15,
// K = kc + (lane>=16 ? 8 : 0) + {0..7} and the same +16.
static __device__ __forceinline__ v16bf load_frag_b(const __bf16* __restrict__ p,
                                                    int kb) {
  v8bf lo = *reinterpret_cast<const v8bf*>(p + kb);
  v8bf hi = *reinterpret_cast<const v8bf*>(p + kb + 16);
  return __builtin_shufflevector(lo, hi, 0,1,2,3,4,5,6,7,8,9,10,11,12,13,14,15);
}

// =====================================================================
// GEMM: C[rows x nout] = act( (A[rows x K] @ W[nout x K]^T + bias) * scale )
// bf16 operands, fp32 accumulate. One wave per 64x32 macro-tile:
// 4 row-tiles x 2 col-tiles = 8 independent WMMAs per K-chunk.
// Hot loop = pure global_load_b128 + v_wmma (no VALU conversions).
//   grid.x = nout/32, grid.y = rows/64, block = 32
// ACT: 0 = none, 1 = softplus.  OUTB: also write bf16 copy of C.
// =====================================================================
template <int ACT, int OUTB>
__global__ __launch_bounds__(32) void gemm_wmma_bf16(
    const __bf16* __restrict__ A, const __bf16* __restrict__ W,
    const float* __restrict__ bias, float* __restrict__ C,
    __bf16* __restrict__ Cb, int K, int nout, float scale)
{
  const int tn   = blockIdx.x;        // 32-wide col block
  const int tm   = blockIdx.y;        // 64-tall row block
  const int lane = threadIdx.x;       // 0..31
  const int lo   = lane & 15;
  const int hi   = lane >> 4;         // 0 or 1

  const __bf16* __restrict__ Ar[4];
  #pragma unroll
  for (int r = 0; r < 4; ++r)
    Ar[r] = A + (size_t)(tm * 64 + r * 16 + lo) * K;

  const __bf16* __restrict__ Wr[2];
  int ncol[2];
  #pragma unroll
  for (int c = 0; c < 2; ++c) {
    ncol[c] = tn * 32 + c * 16 + lo;
    Wr[c] = W + (size_t)ncol[c] * K;
  }

  v8f acc[4][2];
  #pragma unroll
  for (int c = 0; c < 2; ++c) {
    float bv = bias ? bias[ncol[c]] : 0.0f;   // same col for all 8 regs
    #pragma unroll
    for (int r = 0; r < 4; ++r)
      #pragma unroll
      for (int j = 0; j < 8; ++j) acc[r][c][j] = bv;
  }

  for (int kc = 0; kc < K; kc += 32) {
    // prefetch the streamed weight operand ~1KB ahead (global_prefetch_b8)
    __builtin_prefetch(Wr[0] + kc + 512, 0, 0);
    __builtin_prefetch(Wr[1] + kc + 512, 0, 0);

    const int ka = kc + hi * 8;
    v16bf af[4];
    #pragma unroll
    for (int r = 0; r < 4; ++r) af[r] = load_frag_a(Ar[r], ka);

    const int kb = kc + hi * 8;
    #pragma unroll
    for (int c = 0; c < 2; ++c) {
      v16bf bf = load_frag_b(Wr[c], kb);
      #pragma unroll
      for (int r = 0; r < 4; ++r) {
        acc[r][c] = __builtin_amdgcn_wmma_f32_16x16x32_bf16(
            /*neg_a=*/false, af[r], /*neg_b=*/false, bf,
            /*c_mod=*/(short)0, acc[r][c],
            /*reuse_a=*/false, /*reuse_b=*/false);
      }
    }
  }

  // C/D layout: VGPR j -> row = base + j + hi*8, col = ncol[c]
  #pragma unroll
  for (int r = 0; r < 4; ++r) {
    #pragma unroll
    for (int c = 0; c < 2; ++c) {
      #pragma unroll
      for (int j = 0; j < 8; ++j) {
        float v = acc[r][c][j] * scale;
        if (ACT == 1) v = (v > 20.0f) ? v : log1pf(expf(v));
        const int rr = tm * 64 + r * 16 + j + hi * 8;
        C[(size_t)rr * nout + ncol[c]] = v;
        if (OUTB) Cb[(size_t)rr * nout + ncol[c]] = f2bf(v);
      }
    }
  }
}

// =====================================================================
// block reduction helper (blockDim == 256, 8 waves)
// =====================================================================
static __device__ __forceinline__ float block_sum(float v, float* sh) {
  #pragma unroll
  for (int o = 16; o; o >>= 1) v += __shfl_xor(v, o);
  const int w = threadIdx.x >> 5;
  __syncthreads();
  if ((threadIdx.x & 31) == 0) sh[w] = v;
  __syncthreads();
  float r = sh[0];
  #pragma unroll
  for (int i = 1; i < 8; ++i) r += sh[i];
  return r;
}

// =====================================================================
// Fused residual update + LayerNorm.
//   if (Yin): X = Yin + dvec * X   (written back)
//   XN = LN(X) * g + b   (f32, for the scan)  and bf16 copy for the GEMMs
// =====================================================================
__global__ __launch_bounds__(256) void resid_ln_kernel(
    float* __restrict__ X, const float* __restrict__ Yin,
    const float* __restrict__ dvec,
    const float* __restrict__ g, const float* __restrict__ bb,
    float* __restrict__ XN, __bf16* __restrict__ XNB)
{
  __shared__ float sh[8];
  const int row = blockIdx.x;
  const int tid = threadIdx.x;
  float* xr = X + (size_t)row * DD;

  float s = 0.0f;
  if (Yin) {
    const float* yr = Yin + (size_t)row * DD;
    for (int d = tid; d < DD; d += 256) {
      float v = fmaf(dvec[d], xr[d], yr[d]);
      xr[d] = v;
      s += v;
    }
  } else {
    for (int d = tid; d < DD; d += 256) s += xr[d];
  }
  s = block_sum(s, sh);
  const float mu = s * (1.0f / DD);

  float vs = 0.0f;
  for (int d = tid; d < DD; d += 256) { float t = xr[d] - mu; vs += t * t; }
  vs = block_sum(vs, sh);
  const float inv = rsqrtf(vs * (1.0f / DD) + 1e-5f);

  float* xo = XN + (size_t)row * DD;
  __bf16* xb = XNB + (size_t)row * DD;
  for (int d = tid; d < DD; d += 256) {
    float v = (xr[d] - mu) * inv * g[d] + bb[d];
    xo[d] = v;
    xb[d] = f2bf(v);
  }
}

// =====================================================================
// SSM selective-scan: one wave per (b,d) channel; state h[64] as
// 2 floats/lane (n = 2*lane, 2*lane+1). Sequential over t.
// =====================================================================
__global__ __launch_bounds__(256) void ssm_scan_kernel(
    const float* __restrict__ DT, const float* __restrict__ XN,
    const float* __restrict__ BIN, const float* __restrict__ CIN,
    const float* __restrict__ ALOG, float* __restrict__ Y)
{
  const int wave = (blockIdx.x * 256 + threadIdx.x) >> 5;   // 0..B*D-1
  const int lane = threadIdx.x & 31;
  const int b = wave / DD;
  const int d = wave % DD;
  const int n0 = lane * 2;

  const float LOG2E = 1.44269504088896f;
  const float a0 = -expf(ALOG[(size_t)d * NN + n0])     * LOG2E;
  const float a1 = -expf(ALOG[(size_t)d * NN + n0 + 1]) * LOG2E;

  float h0 = 0.0f, h1 = 0.0f;
  const float* dtp = DT + (size_t)b * LL * DD + d;
  const float* xp  = XN + (size_t)b * LL * DD + d;
  const float2* Bp = reinterpret_cast<const float2*>(BIN + (size_t)b * LL * NN);
  const float2* Cp = reinterpret_cast<const float2*>(CIN + (size_t)b * LL * NN);
  float* yp = Y + (size_t)b * LL * DD + d;

  for (int t0 = 0; t0 < LL; t0 += 32) {
    float dtv = dtp[(size_t)(t0 + lane) * DD];
    float xv  = xp [(size_t)(t0 + lane) * DD];
    #pragma unroll 4
    for (int tt = 0; tt < 32; ++tt) {
      const int t = t0 + tt;
      const float dts = __shfl(dtv, tt);
      const float xs  = __shfl(xv,  tt);
      const float2 Bv = Bp[t * 32 + lane];
      const float2 Cv = Cp[t * 32 + lane];
      const float dtx = dts * xs;
      h0 = __builtin_fmaf(h0, exp2f(dts * a0), dtx * Bv.x);
      h1 = __builtin_fmaf(h1, exp2f(dts * a1), dtx * Bv.y);
      float p = __builtin_fmaf(Cv.x, h0, Cv.y * h1);
      #pragma unroll
      for (int o = 16; o; o >>= 1) p += __shfl_xor(p, o);
      if (lane == 0) yp[(size_t)t * DD] = p;
    }
  }
}

// =====================================================================
// mean over L:  Mout[b,d] = mean_t XF[b,t,d]
// =====================================================================
__global__ __launch_bounds__(256) void mean_kernel(
    const float* __restrict__ XF, float* __restrict__ Mout)
{
  const int idx = blockIdx.x * 256 + threadIdx.x;   // b*D + d
  const int b = idx / DD, d = idx % DD;
  const float* p = XF + (size_t)b * LL * DD + d;
  float s = 0.0f;
  for (int t = 0; t < LL; ++t) s += p[(size_t)t * DD];
  Mout[idx] = s * (1.0f / LL);
}

// cond layer 1: silu(m @ w1^T + b1) -> C1 ; one wave per (b,j)
__global__ __launch_bounds__(256) void cond1_kernel(
    const float* __restrict__ Mb, const float* __restrict__ W1,
    const float* __restrict__ B1, float* __restrict__ C1)
{
  const int wave = (blockIdx.x * 256 + threadIdx.x) >> 5;  // 0..511
  const int lane = threadIdx.x & 31;
  const int b = wave >> 8, j = wave & 255;
  const float* m = Mb + (size_t)b * DD;
  const float* w = W1 + (size_t)j * DD;
  float acc = 0.0f;
  for (int k = lane; k < DD; k += 32) acc += m[k] * w[k];
  #pragma unroll
  for (int o = 16; o; o >>= 1) acc += __shfl_xor(acc, o);
  if (lane == 0) {
    float v = acc + B1[j];
    C1[wave] = v / (1.0f + expf(-v));   // silu
  }
}

// cond layer 2: C1 @ w2^T + b2 -> OUT ; one wave per (b,i)
__global__ __launch_bounds__(256) void cond2_kernel(
    const float* __restrict__ C1, const float* __restrict__ W2,
    const float* __restrict__ B2, float* __restrict__ OUT)
{
  const int wave = (blockIdx.x * 256 + threadIdx.x) >> 5;
  const int lane = threadIdx.x & 31;
  const int b = wave >> 8, i = wave & 255;
  const float* c = C1 + (size_t)b * CC;
  const float* w = W2 + (size_t)i * CC;
  float acc = 0.0f;
  for (int k = lane; k < CC; k += 32) acc += c[k] * w[k];
  #pragma unroll
  for (int o = 16; o; o >>= 1) acc += __shfl_xor(acc, o);
  if (lane == 0) OUT[wave] = acc + B2[i];
}

// softmax over last dim (len 256); one block per row, in-place
__global__ __launch_bounds__(256) void softmax_kernel(float* __restrict__ S)
{
  __shared__ float sh[8];
  const int row = blockIdx.x;
  const int tid = threadIdx.x;
  float* p = S + (size_t)row * LL;
  const float v = p[tid];
  float m = v;
  #pragma unroll
  for (int o = 16; o; o >>= 1) m = fmaxf(m, __shfl_xor(m, o));
  if ((tid & 31) == 0) sh[tid >> 5] = m;
  __syncthreads();
  m = sh[0];
  #pragma unroll
  for (int i = 1; i < 8; ++i) m = fmaxf(m, sh[i]);
  __syncthreads();
  const float e = expf(v - m);
  const float s = block_sum(e, sh);
  p[tid] = e / s;
}

// z0[b,m,d] = sum_l attn[b,m,l] * V[b,l,d]    grid (D/256, B*M)
__global__ __launch_bounds__(256) void attnv_kernel(
    const float* __restrict__ ATT, const float* __restrict__ V,
    float* __restrict__ OUT)
{
  const int d  = blockIdx.x * 256 + threadIdx.x;
  const int bm = blockIdx.y;          // b*M + m
  const int b  = bm >> 6;
  const float* arow = ATT + (size_t)bm * LL;
  const float* vp   = V + (size_t)b * LL * DD + d;
  float acc = 0.0f;
  for (int l = 0; l < LL; ++l) acc += arow[l] * vp[(size_t)l * DD];
  OUT[(size_t)bm * DD + d] = acc;
}

// =====================================================================
// Host-side orchestration
// =====================================================================
extern "C" void kernel_launch(void* const* d_in, const int* in_sizes, int n_in,
                              void* d_out, int out_size, void* d_ws, size_t ws_size,
                              hipStream_t stream) {
  const float* emb   = (const float*)d_in[0];
  const float* in_w  = (const float*)d_in[1];
  const float* in_b  = (const float*)d_in[2];
  const float* ng    = (const float*)d_in[3];
  const float* nb    = (const float*)d_in[4];
  const float* dtw   = (const float*)d_in[5];
  const float* dtb   = (const float*)d_in[6];
  const float* Bw    = (const float*)d_in[7];
  const float* Cw    = (const float*)d_in[8];
  const float* Dvec  = (const float*)d_in[9];
  const float* Alog  = (const float*)d_in[10];
  const float* outg  = (const float*)d_in[11];
  const float* outb  = (const float*)d_in[12];
  const float* latq  = (const float*)d_in[13];
  const float* pkw   = (const float*)d_in[14];
  const float* pvw   = (const float*)d_in[15];
  const float* w1    = (const float*)d_in[16];
  const float* b1c   = (const float*)d_in[17];
  const float* w2    = (const float*)d_in[18];
  const float* b2c   = (const float*)d_in[19];

  float* ws = (float*)d_ws;
  constexpr size_t SZ = (size_t)BB * LL * DD;          // 1,835,008 floats
  float* X   = ws;                                     // residual stream
  float* XN  = ws + 1 * SZ;                            // normalized (f32)
  float* DT  = ws + 2 * SZ;                            // dt / later KEYS
  float* Yb  = ws + 3 * SZ;                            // scan out / later VALS
  float* BIN = ws + 4 * SZ;                            // (B,L,N)
  float* CIN = BIN + (size_t)BB * LL * NN;
  float* SC  = CIN + (size_t)BB * LL * NN;             // scores/attn (B,M,L)
  float* Mb  = SC + (size_t)BB * MM * LL;              // (B,D)
  float* C1  = Mb + (size_t)BB * DD;                   // (B,COND)

  // bf16 staging region (after f32 buffers, 512KB-aligned start)
  __bf16* ABF = (__bf16*)(ws + 4 * SZ + 131072);       // activations bf16 (B*L*D)
  __bf16* KBF = ABF + SZ;                              // keys bf16 (B*L*D)
  __bf16* WBF = KBF + SZ;                              // weight bf16 (D*D)

  float* z0_out = (float*)d_out;                       // (B,M,D)
  float* c_out  = (float*)d_out + (size_t)BB * MM * DD;

  const dim3 gBig(DD / 32, (BB * LL) / 64);            // 112 x 8
  const dim3 gSmall(NN / 32, (BB * LL) / 64);          // 2 x 8
  const dim3 gScore(LL / 32, MM / 64);                 // 8 x 1
  const int cvW  = (DD * DD) / 2048;                   // 6272 blocks
  const int cvA  = (int)(SZ / 2048);                   // 896 blocks
  const int cvBC = (NN * DD) / 2048;                   // 112 blocks
  const int cvQ  = (MM * DD) / 2048;                   // 112 blocks

  // ---- input projection: x = emb @ Win^T + b  (both operands -> bf16 once)
  cvt_f32_bf16_kernel<<<cvA, 256, 0, stream>>>(emb, ABF);
  cvt_f32_bf16_kernel<<<cvW, 256, 0, stream>>>(in_w, WBF);
  gemm_wmma_bf16<0,0><<<gBig, 32, 0, stream>>>(ABF, WBF, in_b, X, nullptr, DD, DD, 1.0f);

  for (int i = 0; i < 2; ++i) {
    // residual fold-in (layer 1 consumes layer 0's scan output) + LN
    // writes XN (f32 for scan) and ABF (bf16 for GEMMs)
    resid_ln_kernel<<<BB * LL, 256, 0, stream>>>(
        X, (i == 0) ? nullptr : Yb, (i == 0) ? nullptr : Dvec /*layer 0 D*/,
        ng + (size_t)i * DD, nb + (size_t)i * DD, XN, ABF);

    // dt = softplus(xn @ dt_w^T + dt_b)
    cvt_f32_bf16_kernel<<<cvW, 256, 0, stream>>>(dtw + (size_t)i * DD * DD, WBF);
    gemm_wmma_bf16<1,0><<<gBig, 32, 0, stream>>>(ABF, WBF, dtb + (size_t)i * DD,
                                                 DT, nullptr, DD, DD, 1.0f);
    // B_in / C_in projections
    cvt_f32_bf16_kernel<<<cvBC, 256, 0, stream>>>(Bw + (size_t)i * NN * DD, WBF);
    gemm_wmma_bf16<0,0><<<gSmall, 32, 0, stream>>>(ABF, WBF, nullptr, BIN, nullptr,
                                                   DD, NN, 1.0f);
    cvt_f32_bf16_kernel<<<cvBC, 256, 0, stream>>>(Cw + (size_t)i * NN * DD, WBF);
    gemm_wmma_bf16<0,0><<<gSmall, 32, 0, stream>>>(ABF, WBF, nullptr, CIN, nullptr,
                                                   DD, NN, 1.0f);

    // fused selective scan (never materializes (B,L,D,N))
    ssm_scan_kernel<<<(BB * DD) / 8, 256, 0, stream>>>(
        DT, XN, BIN, CIN, Alog + (size_t)i * DD * NN, Yb);
  }

  // final residual + out-LN -> XN (f32) + ABF (bf16 final features)
  resid_ln_kernel<<<BB * LL, 256, 0, stream>>>(
      X, Yb, Dvec + (size_t)1 * DD, outg, outb, XN, ABF);

  // conditioning path (scalar-heavy, f32)
  mean_kernel<<<(BB * DD) / 256, 256, 0, stream>>>(XN, Mb);
  cond1_kernel<<<(BB * CC) / 8, 256, 0, stream>>>(Mb, w1, b1c, C1);
  cond2_kernel<<<(BB * CC) / 8, 256, 0, stream>>>(C1, w2, b2c, c_out);

  // pooling path: keys (f32 + bf16 shadow) and values
  cvt_f32_bf16_kernel<<<cvW, 256, 0, stream>>>(pkw, WBF);
  gemm_wmma_bf16<0,1><<<gBig, 32, 0, stream>>>(ABF, WBF, nullptr, DT, KBF, DD, DD, 1.0f);
  cvt_f32_bf16_kernel<<<cvW, 256, 0, stream>>>(pvw, WBF);
  gemm_wmma_bf16<0,0><<<gBig, 32, 0, stream>>>(ABF, WBF, nullptr, Yb, nullptr, DD, DD, 1.0f);

  // scores[b] = q @ keys[b]^T / sqrt(D)  (q -> bf16 into ABF head; XN no longer
  // needed as GEMM input after the values GEMM above)
  cvt_f32_bf16_kernel<<<cvQ, 256, 0, stream>>>(latq, ABF);
  const float invs = 1.0f / sqrtf((float)DD);
  for (int b = 0; b < BB; ++b) {
    gemm_wmma_bf16<0,0><<<gScore, 32, 0, stream>>>(
        ABF, KBF + (size_t)b * LL * DD, nullptr,
        SC + (size_t)b * MM * LL, nullptr, DD, LL, invs);
  }
  softmax_kernel<<<BB * MM, 256, 0, stream>>>(SC);
  attnv_kernel<<<dim3(DD / 256, BB * MM), 256, 0, stream>>>(SC, Yb, z0_out);
}
